// AttentionBlock_58445914964576
// MI455X (gfx1250) — matrix operand order repaired and checked by reading, compiled, not run on hardware
//
#include <hip/hip_runtime.h>

// ---------------------------------------------------------------- types
typedef __bf16 bf16;
typedef bf16  bf16x8 __attribute__((ext_vector_type(8)));
typedef bf16  v16bf  __attribute__((ext_vector_type(16)));
typedef float v8f    __attribute__((ext_vector_type(8)));

union FragA { v16bf v; bf16x8 h[2]; };

// async copy: 16B global -> LDS, tracked by ASYNCcnt (CDNA5)
__device__ __forceinline__ void async_b128(unsigned lds_off, const void* gptr) {
  asm volatile("global_load_async_to_lds_b128 %0, %1, off"
               :: "v"(lds_off), "v"((unsigned long long)(size_t)gptr) : "memory");
}
#define LDS_OFF(p) ((unsigned)(size_t)(p))

// ---------------------------------------------------------------- sizes
#define SEQ    2048
#define HIDDEN 2048
#define QKV    6144
#define NH     16
#define HD     128

#define TM 128
#define TN 128
#define TK 32
#define AST 40   // padded LDS row stride (elements) -> 80B, 16B aligned

// ================================================================ f32 -> bf16
__global__ __launch_bounds__(256) void cvt_bf16_kernel(const float* __restrict__ in,
                                                       bf16* __restrict__ out, size_t n) {
  size_t i = (size_t)blockIdx.x * 256 + threadIdx.x;
  size_t stride = (size_t)gridDim.x * 256;
  for (; i < n; i += stride) out[i] = (bf16)in[i];
}

// ================================================================ RMSNorm -> bf16
__global__ __launch_bounds__(256) void rmsnorm_kernel(const float* __restrict__ x,
                                                      const float* __restrict__ scale,
                                                      bf16* __restrict__ t) {
  __shared__ float red[8];
  const int row = blockIdx.x;
  const float* xr = x + (size_t)row * HIDDEN;
  float s = 0.f;
  for (int i = threadIdx.x; i < HIDDEN; i += 256) { float v = xr[i]; s += v * v; }
  for (int m = 16; m >= 1; m >>= 1) s += __shfl_xor(s, m, 32);
  if ((threadIdx.x & 31) == 0) red[threadIdx.x >> 5] = s;
  __syncthreads();
  float tot = 0.f;
  for (int w = 0; w < 8; ++w) tot += red[w];
  const float inv = rsqrtf(tot / (float)HIDDEN + 1e-5f);
  bf16* tr = t + (size_t)row * HIDDEN;
  for (int i = threadIdx.x; i < HIDDEN; i += 256) tr[i] = (bf16)(xr[i] * inv * scale[i]);
}

// ================================================================ NT GEMM (bf16 WMMA)
// C[M,N] = A[M,K] * B[N,K]^T (+bias) ; FINAL: +residual -> f32 out, else -> bf16 out
// Double-buffered LDS tiles filled by async global->LDS copies (ASYNCcnt).
template <bool FINAL>
__global__ __launch_bounds__(256) void gemm_nt_kernel(const bf16* __restrict__ A,
                                                      const bf16* __restrict__ B,
                                                      const float* __restrict__ bias,
                                                      const float* __restrict__ resid,
                                                      float* __restrict__ outF,
                                                      bf16* __restrict__ outB,
                                                      int M, int N, int K) {
  __shared__ __align__(16) bf16 As[2][TM][AST];
  __shared__ __align__(16) bf16 Bs[2][TN][AST];

  const int tid = threadIdx.x;
  const int lane = tid & 31, wave = tid >> 5;
  const int waveM = wave & 3, waveN = wave >> 2;      // 4 x 2 wave grid
  const int m0 = blockIdx.y * TM, n0 = blockIdx.x * TN;
  const int half = lane >> 4, l16 = lane & 15;

  const int lr = tid >> 2;          // 0..63
  const int lc = (tid & 3) * 8;     // 0,8,16,24

  const bf16* Ap0 = A + (size_t)(m0 + lr) * K + lc;
  const bf16* Ap1 = A + (size_t)(m0 + lr + 64) * K + lc;
  const bf16* Bp0 = B + (size_t)(n0 + lr) * K + lc;
  const bf16* Bp1 = B + (size_t)(n0 + lr + 64) * K + lc;

  v8f acc[2][4] = {};
  const int nk = K / TK;

  // prologue: fill buffer 0
  async_b128(LDS_OFF(&As[0][lr][lc]),      Ap0);
  async_b128(LDS_OFF(&As[0][lr + 64][lc]), Ap1);
  async_b128(LDS_OFF(&Bs[0][lr][lc]),      Bp0);
  async_b128(LDS_OFF(&Bs[0][lr + 64][lc]), Bp1);

  for (int kt = 0; kt < nk; ++kt) {
    const int buf = kt & 1;
    if (kt + 1 < nk) {             // overlap next K-slab fetch with WMMA
      const int k1 = (kt + 1) * TK;
      async_b128(LDS_OFF(&As[buf ^ 1][lr][lc]),      Ap0 + k1);
      async_b128(LDS_OFF(&As[buf ^ 1][lr + 64][lc]), Ap1 + k1);
      async_b128(LDS_OFF(&Bs[buf ^ 1][lr][lc]),      Bp0 + k1);
      async_b128(LDS_OFF(&Bs[buf ^ 1][lr + 64][lc]), Bp1 + k1);
      asm volatile("s_wait_asynccnt 0x4" ::: "memory");  // tile kt done (in-order)
    } else {
      asm volatile("s_wait_asynccnt 0x0" ::: "memory");
    }
    __syncthreads();

    FragA af[2], bfr[4];
    for (int mi = 0; mi < 2; ++mi) {
      const int r = waveM * 32 + mi * 16 + l16;       // A: lane = M%16, half = K-subset
      af[mi].h[0] = *(const bf16x8*)&As[buf][r][half * 8];
      af[mi].h[1] = *(const bf16x8*)&As[buf][r][16 + half * 8];
    }
    for (int ni = 0; ni < 4; ++ni) {
      const int c = waveN * 64 + ni * 16 + l16;       // B: lane = N, half = K-subset
      bfr[ni].h[0] = *(const bf16x8*)&Bs[buf][c][half * 8];
      bfr[ni].h[1] = *(const bf16x8*)&Bs[buf][c][16 + half * 8];
    }
    for (int mi = 0; mi < 2; ++mi)
      for (int ni = 0; ni < 4; ++ni)
        acc[mi][ni] = __builtin_amdgcn_wmma_f32_16x16x32_bf16(
            false, af[mi].v, false, bfr[ni].v, (short)0, acc[mi][ni], false, false);
    __syncthreads();   // all waves done reading buf before it is refilled
  }

  for (int mi = 0; mi < 2; ++mi)
    for (int ni = 0; ni < 4; ++ni) {
      const int col = n0 + waveN * 64 + ni * 16 + l16;
      const float bv = bias ? bias[col] : 0.f;
      for (int r = 0; r < 8; ++r) {
        const int row = m0 + waveM * 32 + mi * 16 + r + half * 8;   // C/D layout
        const float v = acc[mi][ni][r] + bv;
        if (FINAL)
          outF[(size_t)row * N + col] = v + resid[(size_t)row * N + col];
        else
          outB[(size_t)row * N + col] = (bf16)v;
      }
    }
}

// ================================================================ RoPE (q,k in-place, bf16)
__global__ __launch_bounds__(256) void rope_kernel(bf16* __restrict__ qkv) {
  const int idx = blockIdx.x * 256 + threadIdx.x;     // SEQ*NH*64 threads
  if (idx >= SEQ * NH * 64) return;
  const int d = idx & 63;
  const int h = (idx >> 6) & (NH - 1);
  const int pos = idx >> 10;
  const float inv = __expf(-((float)(2 * d) / 128.f) * __logf(150000.f));
  const float ang = (float)pos * inv;
  const float c = __cosf(ang), s = __sinf(ang);
  size_t i1 = (size_t)pos * QKV + h * HD + d, i2 = i1 + 64;
  float a = (float)qkv[i1], b = (float)qkv[i2];
  qkv[i1] = (bf16)(a * c - b * s);
  qkv[i2] = (bf16)(b * c + a * s);
  i1 += HIDDEN; i2 += HIDDEN;                          // k section
  a = (float)qkv[i1]; b = (float)qkv[i2];
  qkv[i1] = (bf16)(a * c - b * s);
  qkv[i2] = (bf16)(b * c + a * s);
}

// ================================================================ flash attention (bf16 WMMA)
#define KT 32
__global__ __launch_bounds__(256) void attn_kernel(const bf16* __restrict__ qkv,
                                                   bf16* __restrict__ o) {
  __shared__ __align__(16) bf16 Ks[KT][136];       // 32 keys x 128 d (padded)
  __shared__ __align__(16) bf16 Vt[HD][AST];       // transposed V: d x key
  __shared__ __align__(16) bf16 Pw[8][16][AST];    // per-wave P scratch

  const int h = blockIdx.y;
  const int qbase = blockIdx.x * 128;
  const int tid = threadIdx.x, lane = tid & 31, wave = tid >> 5;
  const int half = lane >> 4, l16 = lane & 15;
  const int qw = qbase + wave * 16;                 // this wave's 16 queries

  // Q fragments straight from global (A layout: lane = M%16)
  FragA qf[4];
  const bf16* qrow = qkv + (size_t)(qw + l16) * QKV + h * HD;
  for (int dt = 0; dt < 4; ++dt) {
    qf[dt].h[0] = *(const bf16x8*)&qrow[dt * 32 + half * 8];
    qf[dt].h[1] = *(const bf16x8*)&qrow[dt * 32 + 16 + half * 8];
  }

  v8f oacc[8] = {};
  float mrow[8], lrow[8];
  for (int r = 0; r < 8; ++r) { mrow[r] = -3.0e38f; lrow[r] = 0.f; }

  const float sc = 0.08838834764831845f;            // 1/sqrt(128)
  const int ntiles = qbase / KT + 4;                // causal: keys 0..qbase+127
  const int krow = tid >> 3;                        // 0..31
  const int kc = (tid & 7) * 16;                    // 0..112

  for (int kt = 0; kt < ntiles; ++kt) {
    const int kbase = kt * KT;
    {
      // K tile: async global -> LDS (ASYNCcnt)
      const bf16* ksrc = qkv + (size_t)(kbase + krow) * QKV + HIDDEN + h * HD + kc;
      async_b128(LDS_OFF(&Ks[krow][kc]),     ksrc);
      async_b128(LDS_OFF(&Ks[krow][kc + 8]), ksrc + 8);
      // V tile transposed via VGPRs + scalar DS stores
      const bf16* vsrc = qkv + (size_t)(kbase + krow) * QKV + 2 * HIDDEN + h * HD + kc;
      bf16x8 v0 = *(const bf16x8*)vsrc;
      bf16x8 v1 = *(const bf16x8*)(vsrc + 8);
      for (int j = 0; j < 8; ++j) { Vt[kc + j][krow] = v0[j]; Vt[kc + 8 + j][krow] = v1[j]; }
      asm volatile("s_wait_asynccnt 0x0" ::: "memory");
    }
    __syncthreads();

    // S = Q K^T  (two 16-key groups)
    v8f s[2] = {};
    for (int kg = 0; kg < 2; ++kg)
      for (int dt = 0; dt < 4; ++dt) {
        FragA kb;
        kb.h[0] = *(const bf16x8*)&Ks[kg * 16 + l16][dt * 32 + half * 8];
        kb.h[1] = *(const bf16x8*)&Ks[kg * 16 + l16][dt * 32 + 16 + half * 8];
        s[kg] = __builtin_amdgcn_wmma_f32_16x16x32_bf16(
            false, qf[dt].v, false, kb.v, (short)0, s[kg], false, false);
      }

    // scale + causal mask + online softmax (reduce over 16-lane N group)
    for (int r = 0; r < 8; ++r) {
      const int qi = qw + r + half * 8;
      float mx = -3.0e38f;
      for (int kg = 0; kg < 2; ++kg) {
        const int kj = kbase + kg * 16 + l16;
        const float v = (kj <= qi) ? s[kg][r] * sc : -3.0e38f;
        s[kg][r] = v;
        mx = fmaxf(mx, v);
      }
      for (int msk = 1; msk < 16; msk <<= 1) mx = fmaxf(mx, __shfl_xor(mx, msk, 32));
      const float mnew = fmaxf(mrow[r], mx);
      const float alpha = __expf(mrow[r] - mnew);
      float rsum = 0.f;
      for (int kg = 0; kg < 2; ++kg) {
        const float p = __expf(s[kg][r] - mnew);
        s[kg][r] = p;
        rsum += p;
      }
      for (int msk = 1; msk < 16; msk <<= 1) rsum += __shfl_xor(rsum, msk, 32);
      lrow[r] = lrow[r] * alpha + rsum;
      mrow[r] = mnew;
      for (int dt8 = 0; dt8 < 8; ++dt8) oacc[dt8][r] *= alpha;
    }

    // P: C-layout -> A-layout via per-wave LDS scratch
    for (int r = 0; r < 8; ++r)
      for (int kg = 0; kg < 2; ++kg)
        Pw[wave][r + half * 8][kg * 16 + l16] = (bf16)s[kg][r];
    asm volatile("s_wait_dscnt 0x0" ::: "memory");   // wave-local LDS RAW fence

    FragA pf;
    pf.h[0] = *(const bf16x8*)&Pw[wave][l16][half * 8];
    pf.h[1] = *(const bf16x8*)&Pw[wave][l16][16 + half * 8];
    for (int dt8 = 0; dt8 < 8; ++dt8) {
      FragA vf;
      vf.h[0] = *(const bf16x8*)&Vt[dt8 * 16 + l16][half * 8];
      vf.h[1] = *(const bf16x8*)&Vt[dt8 * 16 + l16][16 + half * 8];
      oacc[dt8] = __builtin_amdgcn_wmma_f32_16x16x32_bf16(
          false, pf.v, false, vf.v, (short)0, oacc[dt8], false, false);
    }
    __syncthreads();
  }

  for (int r = 0; r < 8; ++r) lrow[r] = 1.f / lrow[r];
  for (int dt8 = 0; dt8 < 8; ++dt8)
    for (int r = 0; r < 8; ++r) {
      const int row = qw + r + half * 8;
      const int col = h * HD + dt8 * 16 + l16;
      o[(size_t)row * HIDDEN + col] = (bf16)(oacc[dt8][r] * lrow[r]);
    }
}

// ================================================================ launch
extern "C" void kernel_launch(void* const* d_in, const int* in_sizes, int n_in,
                              void* d_out, int out_size, void* d_ws, size_t ws_size,
                              hipStream_t stream) {
  const float* x          = (const float*)d_in[0];
  const float* norm_scale = (const float*)d_in[1];
  const float* qkv_w      = (const float*)d_in[2];
  const float* qkv_b      = (const float*)d_in[3];
  const float* out_w      = (const float*)d_in[4];
  const float* out_b      = (const float*)d_in[5];
  float* out = (float*)d_out;

  // workspace layout (bf16 buffers), total 72 MB
  char* ws = (char*)d_ws;
  if (ws_size < (size_t)72 * 1024 * 1024) return;
  bf16* t_bf    = (bf16*)(ws);                          //  8 MB: rmsnorm(x)
  bf16* qkvw_bf = (bf16*)(ws + ((size_t)8  << 20));     // 24 MB
  bf16* outw_bf = (bf16*)(ws + ((size_t)32 << 20));     //  8 MB
  bf16* qkv_bf  = (bf16*)(ws + ((size_t)40 << 20));     // 24 MB: q|k|v
  bf16* o_bf    = (bf16*)(ws + ((size_t)64 << 20));     //  8 MB: attn out

  // 1) weight conversion + rmsnorm
  cvt_bf16_kernel<<<4096, 256, 0, stream>>>(qkv_w, qkvw_bf, (size_t)QKV * HIDDEN);
  cvt_bf16_kernel<<<2048, 256, 0, stream>>>(out_w, outw_bf, (size_t)HIDDEN * HIDDEN);
  rmsnorm_kernel<<<SEQ, 256, 0, stream>>>(x, norm_scale, t_bf);

  // 2) QKV = t @ qkv_w^T + b   -> bf16
  gemm_nt_kernel<false><<<dim3(QKV / TN, SEQ / TM), 256, 0, stream>>>(
      t_bf, qkvw_bf, qkv_b, nullptr, nullptr, qkv_bf, SEQ, QKV, HIDDEN);

  // 3) RoPE on q,k
  rope_kernel<<<(SEQ * NH * 64) / 256, 256, 0, stream>>>(qkv_bf);

  // 4) flash attention -> o_bf
  attn_kernel<<<dim3(SEQ / 128, NH), 256, 0, stream>>>(qkv_bf, o_bf);

  // 5) out = x + o @ out_w^T + out_b   -> f32
  gemm_nt_kernel<true><<<dim3(HIDDEN / TN, SEQ / TM), 256, 0, stream>>>(
      o_bf, outw_bf, out_b, x, out, nullptr, SEQ, HIDDEN, HIDDEN);
}
